// APEX_MoE_Planner_88905823027289
// MI455X (gfx1250) — compile-verified, matrix operand order checked by reading, not run on hardware
//
#include <hip/hip_runtime.h>
#include <hip/hip_bf16.h>
#include <math.h>

// ---------------------------------------------------------------------------
// MI455X (gfx1250) implementation. wave32, WMMA bf16 16x16x32, f32 accumulate.
// ---------------------------------------------------------------------------

typedef __bf16 bf16;
typedef __attribute__((ext_vector_type(16))) __bf16 v16bf;
typedef __attribute__((ext_vector_type(8)))  float  v8f;

union BF16x16 { v16bf v; uint4 u[2]; };

static __device__ __forceinline__ bf16 f2bf(float x) { return (bf16)x; }

// Hardware tanh (gfx1250 TRANS32 op) -> branch-free gelu.
static __device__ __forceinline__ float tanh_hw(float x) {
#if __has_builtin(__builtin_amdgcn_tanhf)
    return __builtin_amdgcn_tanhf(x);
#else
    float r;
    asm("v_tanh_f32 %0, %1" : "=v"(r) : "v"(x));
    return r;
#endif
}

static __device__ __forceinline__ float gelu_f(float x) {
    // jax.nn.gelu default (tanh approximation)
    // u = sqrt(2/pi) * (x + 0.044715 x^3) = 0.79788456*x + 0.035677408*x^3
    float u = x * (0.7978845608028654f + 0.03567740813636141f * x * x);
    return 0.5f * x * (1.0f + tanh_hw(u));
}

static __device__ __forceinline__ v8f wmma_bf16(v16bf a, v16bf b, v8f c) {
    return __builtin_amdgcn_wmma_f32_16x16x32_bf16(
        /*neg_a=*/false, a, /*neg_b=*/false, b,
        /*c_mod=*/(short)0, c, /*reuse_a=*/false, /*reuse_b=*/false);
}

// A operand (16x32 bf16, MxK) from a row-major LDS tile (16 rows x rowK bf16).
// Per ISA layout: lanes 0-15 -> M=lane, K = [0..7, 16..23]+kbase
//                 lanes16-31 -> M=lane-16, K = [8..15, 24..31]+kbase
static __device__ __forceinline__ v16bf load_a_lds(const bf16* tile, int rowK,
                                                   int lane, int kbase) {
    int m = lane & 15, kh = lane >> 4;
    const bf16* p = tile + m * rowK + kbase + kh * 8;
    BF16x16 r;
    r.u[0] = *(const uint4*)(p);
    r.u[1] = *(const uint4*)(p + 16);
    return r.v;
}

// B operand from pre-swizzled global weights: per (ktile,ntile,lane) 16 bf16
// contiguous (lane holds column n = nt*16 + lane%16, K-halves by lane>>4).
static __device__ __forceinline__ v16bf load_b_packed(const bf16* base, int ntiles,
                                                      int kt, int nt, int lane) {
    const bf16* p = base + (((size_t)(kt * ntiles + nt)) * 32 + lane) * 16;
    BF16x16 r;
    r.u[0] = *(const uint4*)(p);
    r.u[1] = *(const uint4*)(p + 8);
    return r.v;
}

// ---------------------------------------------------------------------------
// Prep kernels
// ---------------------------------------------------------------------------

__global__ void abar_kernel(float* abar) {
    if (blockIdx.x == 0 && threadIdx.x == 0) {
        float cum = 1.0f;
        for (int i = 0; i < 1000; i++) {
            float beta = 0.0001f + (0.02f - 0.0001f) * (float)i / 999.0f;
            cum *= (1.0f - beta);
            abar[i] = cum;
        }
    }
}

// Pack row-major f32 weight (K x N) into bf16 WMMA-B operand order.
__global__ void pack_b_kernel(const float* __restrict__ src, bf16* __restrict__ dst,
                              int K, int N) {
    int idx = blockIdx.x * blockDim.x + threadIdx.x;   // one per (tile,lane)
    int total = (K / 32) * (N / 16) * 32;
    if (idx >= total) return;
    int lane = idx & 31, tile = idx >> 5;
    int ntiles = N / 16;
    int kt = tile / ntiles, nt = tile % ntiles;
    int n = nt * 16 + (lane & 15);
    int kb = kt * 32 + (lane >> 4) * 8;
    bf16* d = dst + (size_t)idx * 16;
    for (int j = 0; j < 8; j++) d[j]     = f2bf(src[(size_t)(kb + j)      * N + n]);
    for (int j = 0; j < 8; j++) d[8 + j] = f2bf(src[(size_t)(kb + 16 + j) * N + n]);
}

// ---------------------------------------------------------------------------
// Encoder: global_ctx[b] = mean_t gelu(obs[b] @ W_enc + stat_proj[b] + b_enc)
// one block per sample, 8 waves, WMMA 256x512 GEMM fused with mean.
// ---------------------------------------------------------------------------

__global__ __launch_bounds__(256)
void encoder_kernel(const float* __restrict__ obs, const float* __restrict__ sc,
                    const float* __restrict__ Wstat, const float* __restrict__ benc,
                    const bf16* __restrict__ WencP, float* __restrict__ gctx) {
    int b = blockIdx.x;
    __shared__ float s_stat[512];
    __shared__ float s_sc[32];
    __shared__ __align__(16) bf16 s_a[16 * 64];

    int tid = threadIdx.x, wave = tid >> 5, lane = tid & 31;
    if (tid < 32) s_sc[tid] = sc[(size_t)b * 32 + tid];
    __syncthreads();
    for (int n = tid; n < 512; n += 256) {
        float acc = benc[n];
        for (int k = 0; k < 32; k++) acc += s_sc[k] * Wstat[k * 512 + n];
        s_stat[n] = acc;
    }

    // Preload B operands (wave owns 4 N-tiles, K = 64 = 2 ktiles)
    v16bf Bop[4][2];
    for (int i = 0; i < 4; i++) {
        int nt = wave * 4 + i;
        Bop[i][0] = load_b_packed(WencP, 32, 0, nt, lane);
        Bop[i][1] = load_b_packed(WencP, 32, 1, nt, lane);
    }
    float colacc[4] = {0.f, 0.f, 0.f, 0.f};
    __syncthreads();

    for (int mt = 0; mt < 16; mt++) {
        // stage A tile (16 rows of obs, K=64), f32 -> bf16
        for (int i = tid; i < 16 * 64; i += 256) {
            int m = i >> 6, k = i & 63;
            s_a[i] = f2bf(obs[((size_t)b * 256 + mt * 16 + m) * 64 + k]);
        }
        // prefetch next M-tile while this one computes (global_prefetch_b8)
        if (mt < 15)
            __builtin_prefetch(&obs[((size_t)b * 256 + (mt + 1) * 16 + (tid >> 4)) * 64 +
                                    (tid & 15) * 4], 0, 1);
        __syncthreads();
        v16bf a0 = load_a_lds(s_a, 64, lane, 0);
        v16bf a1 = load_a_lds(s_a, 64, lane, 32);
        for (int i = 0; i < 4; i++) {
            int nt = wave * 4 + i;
            v8f c = {};
            c = wmma_bf16(a0, Bop[i][0], c);
            c = wmma_bf16(a1, Bop[i][1], c);
            int n = nt * 16 + (lane & 15);
            float sb = s_stat[n];
            float s = 0.f;
            for (int r = 0; r < 8; r++) s += gelu_f(c[r] + sb);
            colacc[i] += s;
        }
        __syncthreads();
    }
    // lane l and l+16 hold the two M-halves of the same column N
    for (int i = 0; i < 4; i++) {
        float other = __shfl_xor(colacc[i], 16);
        if (lane < 16) {
            int n = (wave * 4 + i) * 16 + lane;
            gctx[(size_t)b * 512 + n] = (colacc[i] + other) * (1.0f / 256.0f);
        }
    }
}

// ---------------------------------------------------------------------------
// Router + selection + sinusoidal time embedding (one block per sample)
// ---------------------------------------------------------------------------

__global__ __launch_bounds__(128)
void router_kernel(const float* __restrict__ gctx, const float* __restrict__ Wr,
                   const float* __restrict__ br, const int* __restrict__ phase,
                   const int* __restrict__ tarr, float* __restrict__ probs,
                   int* __restrict__ selected, float* __restrict__ temb) {
    int b = blockIdx.x, tid = threadIdx.x;
    __shared__ float red[6][128];
    float loc[6] = {0.f, 0.f, 0.f, 0.f, 0.f, 0.f};
    for (int k = tid; k < 512; k += 128) {
        float g = gctx[(size_t)b * 512 + k];
        for (int e = 0; e < 6; e++) loc[e] += g * Wr[k * 6 + e];
    }
    for (int e = 0; e < 6; e++) red[e][tid] = loc[e];
    __syncthreads();
    if (tid == 0) {
        float logits[6];
        for (int e = 0; e < 6; e++) {
            float s = 0.f;
            for (int i = 0; i < 128; i++) s += red[e][i];
            logits[e] = s + br[e];
        }
        float m = logits[0];
        for (int e = 1; e < 6; e++) m = fmaxf(m, logits[e]);
        float ex[6], se = 0.f;
        for (int e = 0; e < 6; e++) { ex[e] = __expf(logits[e] - m); se += ex[e]; }
        for (int e = 0; e < 6; e++) probs[(size_t)b * 6 + e] = ex[e] / se;
        int p = phase[b];
        selected[b] = p + ((ex[p + 3] > ex[p]) ? 3 : 0);
    }
    int tb = tarr[b];
    for (int i = tid; i < 256; i += 128) {
        float freq = __expf(-logf(10000.0f) * (float)i / 256.0f);
        float a = (float)tb * freq;
        temb[(size_t)b * 512 + i]       = __sinf(a);
        temb[(size_t)b * 512 + 256 + i] = __cosf(a);
    }
}

// ---------------------------------------------------------------------------
// h-bias: hb[e][b][h] = [gctx(b)|temb(b)] @ W1[e][64:1088] + b1[e]
// grid = 6 experts * 32 sample-tiles; M=16 samples, N=512, K=1024 WMMA GEMM.
// ---------------------------------------------------------------------------

__global__ __launch_bounds__(256)
void hbias_kernel(const float* __restrict__ gctx, const float* __restrict__ temb,
                  const bf16* __restrict__ W1ctP, const float* __restrict__ b1,
                  float* __restrict__ hb) {
    int e = blockIdx.x / 32, mt = blockIdx.x % 32;
    __shared__ __align__(16) bf16 s_a[16 * 32];
    int tid = threadIdx.x, wave = tid >> 5, lane = tid & 31;
    const bf16* Bbase = W1ctP + (size_t)e * 1024 * 512;

    v8f c[4];
    for (int i = 0; i < 4; i++) c[i] = (v8f){};
    for (int ks = 0; ks < 32; ks++) {
        __syncthreads();
        for (int i = tid; i < 16 * 32; i += 256) {
            int m = i >> 5, k = i & 31;
            int bidx = mt * 16 + m, kk = ks * 32 + k;
            float v = (kk < 512) ? gctx[(size_t)bidx * 512 + kk]
                                 : temb[(size_t)bidx * 512 + (kk - 512)];
            s_a[i] = f2bf(v);
        }
        __syncthreads();
        v16bf a = load_a_lds(s_a, 32, lane, 0);
        for (int i = 0; i < 4; i++) {
            int nt = wave * 4 + i;
            c[i] = wmma_bf16(a, load_b_packed(Bbase, 32, ks, nt, lane), c[i]);
        }
    }
    for (int i = 0; i < 4; i++) {
        int nt = wave * 4 + i, n = nt * 16 + (lane & 15);
        float bias = b1[e * 512 + n];
        int mbase = mt * 16 + (lane >> 4) * 8;
        for (int r = 0; r < 8; r++)
            hb[((size_t)e * 512 + (mbase + r)) * 512 + n] = c[i][r] + bias;
    }
}

// ---------------------------------------------------------------------------
// Fused expert tile: noisy_fut -> layer1 (C init = h-bias) -> gelu -> layer2.
// 128 threads (4 waves). Output: each wave holds one 16x16 f32 tile (f range).
// ---------------------------------------------------------------------------

static __device__ __forceinline__ void expert_tile(
    int e, int b, int mt,
    const float* __restrict__ fut, const float* __restrict__ eps,
    const float* __restrict__ abar, const int* __restrict__ tarr,
    const bf16* __restrict__ W1nfP, const bf16* __restrict__ W2P,
    const float* __restrict__ hb, const float* __restrict__ b2,
    bf16* s_a, bf16* s_h, v8f& oc) {
    const int tid = threadIdx.x, wave = tid >> 5, lane = tid & 31;
    int tb = tarr[b];
    float ab = abar[tb];
    float sa = sqrtf(ab), sn = sqrtf(1.0f - ab);

    for (int i = tid; i < 16 * 64; i += 128) {
        int m = i >> 6, k = i & 63;
        size_t g = ((size_t)b * 64 + mt * 16 + m) * 64 + k;
        s_a[i] = f2bf(sa * fut[g] + sn * eps[g]);
    }
    __syncthreads();

    v16bf a0 = load_a_lds(s_a, 64, lane, 0);
    v16bf a1 = load_a_lds(s_a, 64, lane, 32);
    const bf16* B1 = W1nfP + (size_t)e * 64 * 512;
    for (int j = 0; j < 8; j++) {
        int nt = wave * 8 + j, n = nt * 16 + (lane & 15);
        float hbv = hb[((size_t)e * 512 + b) * 512 + n];
        v8f c;
        for (int r = 0; r < 8; r++) c[r] = hbv;
        c = wmma_bf16(a0, load_b_packed(B1, 32, 0, nt, lane), c);
        c = wmma_bf16(a1, load_b_packed(B1, 32, 1, nt, lane), c);
        int mb = (lane >> 4) * 8;
        for (int r = 0; r < 8; r++) s_h[(mb + r) * 512 + n] = f2bf(gelu_f(c[r]));
    }
    __syncthreads();

    const bf16* B2 = W2P + (size_t)e * 512 * 64;
    int nt = wave, n = nt * 16 + (lane & 15);
    float b2v = b2[e * 64 + n];
    v8f c;
    for (int r = 0; r < 8; r++) c[r] = b2v;
    for (int ks = 0; ks < 16; ks++) {
        v16bf a = load_a_lds(s_h, 512, lane, ks * 32);
        c = wmma_bf16(a, load_b_packed(B2, 4, ks, nt, lane), c);
    }
    oc = c;
}

// Selected path: per (b, t-tile) block -> weighted MSE partial (deterministic).
__global__ __launch_bounds__(128)
void selected_kernel(const float* __restrict__ fut, const float* __restrict__ eps,
                     const float* __restrict__ abar, const int* __restrict__ tarr,
                     const bf16* __restrict__ W1nfP, const bf16* __restrict__ W2P,
                     const float* __restrict__ hb, const float* __restrict__ b2,
                     const int* __restrict__ selected, float* __restrict__ partials) {
    int b = blockIdx.x >> 2, mt = blockIdx.x & 3;
    int e = selected[b];
    __shared__ __align__(16) bf16 s_a[16 * 64];
    __shared__ __align__(16) bf16 s_h[16 * 512];
    __shared__ float s_red[128];
    v8f oc;
    expert_tile(e, b, mt, fut, eps, abar, tarr, W1nfP, W2P, hb, b2, s_a, s_h, oc);

    int tid = threadIdx.x, wave = tid >> 5, lane = tid & 31;
    int n = wave * 16 + (lane & 15);
    int mb = (lane >> 4) * 8;
    float p = 0.f;
    for (int r = 0; r < 8; r++) {
        int t = mt * 16 + mb + r;
        float d = oc[r] - eps[((size_t)b * 64 + t) * 64 + n];
        p += d * d;
    }
    s_red[tid] = p;
    __syncthreads();
    if (tid == 0) {
        float s = 0.f;
        for (int i = 0; i < 128; i++) s += s_red[i];
        float w = (e % 3 == 0) ? 1.0f : 5.0f;   // PHASE_WEIGHTS
        partials[blockIdx.x] = s * w * (1.0f / (64.0f * 64.0f * 512.0f));
    }
}

// Regularizer path: outputs_all for all experts, first 4 samples.
__global__ __launch_bounds__(128)
void outsub_kernel(const float* __restrict__ fut, const float* __restrict__ eps,
                   const float* __restrict__ abar, const int* __restrict__ tarr,
                   const bf16* __restrict__ W1nfP, const bf16* __restrict__ W2P,
                   const float* __restrict__ hb, const float* __restrict__ b2,
                   float* __restrict__ outsub) {
    int e = blockIdx.x / 16, rem = blockIdx.x % 16;
    int s = rem >> 2, mt = rem & 3;
    __shared__ __align__(16) bf16 s_a[16 * 64];
    __shared__ __align__(16) bf16 s_h[16 * 512];
    v8f oc;
    expert_tile(e, s, mt, fut, eps, abar, tarr, W1nfP, W2P, hb, b2, s_a, s_h, oc);

    int tid = threadIdx.x, wave = tid >> 5, lane = tid & 31;
    int n = wave * 16 + (lane & 15);
    int mb = (lane >> 4) * 8;
    for (int r = 0; r < 8; r++) {
        int t = mt * 16 + mb + r;
        outsub[(((size_t)e * 4 + s) * 64 + t) * 64 + n] = oc[r];
    }
}

// chain + smooth regularizers over outsub (6,4,64,64); single block, fixed order.
__global__ __launch_bounds__(256)
void reg_kernel(const float* __restrict__ outsub, float* __restrict__ scal) {
    int tid = threadIdx.x;
    const int pi[7] = {0, 1, 3, 4, 0, 1, 2};
    const int pj[7] = {1, 2, 4, 5, 3, 4, 5};
    float chain = 0.f;
    for (int pr = 0; pr < 7; pr++) {
        const float* A = outsub + (size_t)pi[pr] * 16384;
        const float* B = outsub + (size_t)pj[pr] * 16384;
        float acc = 0.f;
        for (int i = tid; i < 16384; i += 256) {
            float d = A[i] - B[i];
            acc += d * d;
        }
        chain += acc * (1.0f / 16384.0f);
    }
    float smooth = 0.f;   // diff along t axis: (E,S,63,64) per-expert mean, summed
    for (int e = 0; e < 6; e++) {
        const float* O = outsub + (size_t)e * 16384;
        float acc = 0.f;
        for (int i = tid; i < 4 * 63 * 64; i += 256) {
            int f = i & 63;
            int tt = (i >> 6) % 63;
            int s = i / (63 * 64);
            float d = O[((size_t)s * 64 + tt + 1) * 64 + f] -
                      O[((size_t)s * 64 + tt) * 64 + f];
            acc += d * d;
        }
        smooth += acc * (1.0f / (4.0f * 63.0f * 64.0f));
    }
    __shared__ float red[256];
    red[tid] = chain + 0.5f * smooth;
    __syncthreads();
    if (tid == 0) {
        float t = 0.f;
        for (int i = 0; i < 256; i++) t += red[i];
        scal[0] = t * 0.01f;   // LAMBDA_REG
    }
}

// main-loss partial sum + load-balance term + final scalar (deterministic).
__global__ __launch_bounds__(256)
void final_kernel(const float* __restrict__ partials, const float* __restrict__ scal,
                  const int* __restrict__ selected, const float* __restrict__ probs,
                  float* __restrict__ out) {
    int tid = threadIdx.x;
    __shared__ float red[256];
    float acc = 0.f;
    for (int i = tid; i < 2048; i += 256) acc += partials[i];
    red[tid] = acc;
    __syncthreads();
    if (tid == 0) {
        float main = 0.f;
        for (int i = 0; i < 256; i++) main += red[i];
        float f[6] = {0, 0, 0, 0, 0, 0}, P[6] = {0, 0, 0, 0, 0, 0};
        for (int b = 0; b < 512; b++) {
            f[selected[b]] += 1.0f;
            for (int e = 0; e < 6; e++) P[e] += probs[(size_t)b * 6 + e];
        }
        float lb = 0.f;
        for (int e = 0; e < 6; e++) lb += (f[e] / 512.0f) * (P[e] / 512.0f);
        out[0] = main + scal[0] + 0.01f * 6.0f * lb;   // LAMBDA_LB * E
    }
}

// ---------------------------------------------------------------------------
// Host launch
// ---------------------------------------------------------------------------

extern "C" void kernel_launch(void* const* d_in, const int* in_sizes, int n_in,
                              void* d_out, int out_size, void* d_ws, size_t ws_size,
                              hipStream_t stream) {
    const float* obs   = (const float*)d_in[0];
    const float* fut   = (const float*)d_in[1];
    const float* sc    = (const float*)d_in[2];
    const float* eps   = (const float*)d_in[3];
    const int*   phase = (const int*)d_in[4];
    const int*   tarr  = (const int*)d_in[5];
    const float* Wenc  = (const float*)d_in[6];
    const float* benc  = (const float*)d_in[7];
    const float* Wstat = (const float*)d_in[8];
    const float* Wr    = (const float*)d_in[9];
    const float* br    = (const float*)d_in[10];
    const float* W1    = (const float*)d_in[11];
    const float* b1    = (const float*)d_in[12];
    const float* W2    = (const float*)d_in[13];
    const float* b2    = (const float*)d_in[14];
    float* out = (float*)d_out;

    size_t off = 0;
    auto alloc = [&](size_t bytes) -> char* {
        size_t o = (off + 255) & ~(size_t)255;
        off = o + bytes;
        return (char*)d_ws + o;
    };
    bf16*  wencP  = (bf16*)alloc((size_t)64 * 512 * 2);
    bf16*  w1nfP  = (bf16*)alloc((size_t)6 * 64 * 512 * 2);
    bf16*  w1ctP  = (bf16*)alloc((size_t)6 * 1024 * 512 * 2);
    bf16*  w2P    = (bf16*)alloc((size_t)6 * 512 * 64 * 2);
    float* gctx   = (float*)alloc((size_t)512 * 512 * 4);
    float* temb   = (float*)alloc((size_t)512 * 512 * 4);
    float* probs  = (float*)alloc((size_t)512 * 6 * 4);
    int*   sel    = (int*)alloc((size_t)512 * 4);
    float* abar   = (float*)alloc((size_t)1000 * 4);
    float* hb     = (float*)alloc((size_t)6 * 512 * 512 * 4);
    float* outsub = (float*)alloc((size_t)6 * 4 * 64 * 64 * 4);
    float* parts  = (float*)alloc((size_t)2048 * 4);
    float* scal   = (float*)alloc((size_t)64);
    (void)in_sizes; (void)n_in; (void)out_size; (void)ws_size;

    auto packGrid = [](int K, int N) {
        int total = (K / 32) * (N / 16) * 32;
        return (total + 255) / 256;
    };

    abar_kernel<<<1, 32, 0, stream>>>(abar);
    pack_b_kernel<<<packGrid(64, 512), 256, 0, stream>>>(Wenc, wencP, 64, 512);
    for (int e = 0; e < 6; e++) {
        const float* W1e = W1 + (size_t)e * 1088 * 512;
        pack_b_kernel<<<packGrid(64, 512), 256, 0, stream>>>(
            W1e, w1nfP + (size_t)e * 64 * 512, 64, 512);
        pack_b_kernel<<<packGrid(1024, 512), 256, 0, stream>>>(
            W1e + (size_t)64 * 512, w1ctP + (size_t)e * 1024 * 512, 1024, 512);
        pack_b_kernel<<<packGrid(512, 64), 256, 0, stream>>>(
            W2 + (size_t)e * 512 * 64, w2P + (size_t)e * 512 * 64, 512, 64);
    }

    encoder_kernel<<<512, 256, 0, stream>>>(obs, sc, Wstat, benc, wencP, gctx);
    router_kernel<<<512, 128, 0, stream>>>(gctx, Wr, br, phase, tarr, probs, sel, temb);
    hbias_kernel<<<192, 256, 0, stream>>>(gctx, temb, w1ctP, b1, hb);
    selected_kernel<<<2048, 128, 0, stream>>>(fut, eps, abar, tarr, w1nfP, w2P,
                                              hb, b2, sel, parts);
    outsub_kernel<<<96, 128, 0, stream>>>(fut, eps, abar, tarr, w1nfP, w2P,
                                          hb, b2, outsub);
    reg_kernel<<<1, 256, 0, stream>>>(outsub, scal);
    final_kernel<<<1, 256, 0, stream>>>(parts, scal, sel, probs, out);
}
